// TensorTwistorDecoder_5660766896355
// MI455X (gfx1250) — matrix operand order repaired and checked by reading, not compiled, not run on hardware
//
#include <hip/hip_runtime.h>
#include <hip/hip_bf16.h>
#include <math.h>

#define HIDDEN 512
#define FF     1024
#define BATCH  128
#define OUTN   512
#define ICHUNK 8
#define BTILE  2048   // 64 cols x 32 k, ushort elements

typedef __attribute__((ext_vector_type(16))) __bf16       v16bf;
typedef __attribute__((ext_vector_type(8)))  float        v8f;
typedef __attribute__((ext_vector_type(4)))  unsigned int v4u;

union BfFrag {
    v16bf v;
    v4u   q[2];
};

__device__ __forceinline__ unsigned short f2bf(float x) {
    // round-to-nearest-even fp32 -> bf16
    unsigned int u = __float_as_uint(x);
    u += 0x7FFFu + ((u >> 16) & 1u);
    return (unsigned short)(u >> 16);
}

// ---------------------------------------------------------------- zero
__global__ void zero_kernel(float* __restrict__ p, int n) {
    int i = blockIdx.x * blockDim.x + threadIdx.x;
    if (i < n) p[i] = 0.0f;
}

// ------------------------------------------------- v = tanh(z@Wv + bv)
__global__ void __launch_bounds__(256)
vproj_kernel(const float* __restrict__ z, const float* __restrict__ Wv,
             const float* __restrict__ bv,
             float* __restrict__ vF, unsigned short* __restrict__ vH) {
    int idx = blockIdx.x * 256 + threadIdx.x;     // 65536 = 128*512
    int b = idx >> 9;
    int o = idx & 511;
    const float* zr = z + b * HIDDEN;
    float acc = bv[o];
#pragma unroll 8
    for (int k = 0; k < HIDDEN; ++k)
        acc += zr[k] * Wv[k * HIDDEN + o];
    float v = tanhf(acc);
    vF[idx] = v;
    vH[idx] = f2bf(v);
}

// stage one W1 tile [32 k x 64 f] as bf16, K-major ([f][kk]) into LDS.
// Thread t loads 8 contiguous floats (two b128 loads) of row kk = t/8.
__device__ __forceinline__ void stage_tile(unsigned short* __restrict__ dst,
                                           const float* __restrict__ W1,
                                           size_t kbase, int tid) {
    const int kk = tid >> 3;
    const int fb = (tid & 7) * 8;
    const float4* src = (const float4*)(W1 + kbase + (size_t)kk * FF + fb);
    float4 a = src[0];
    float4 b = src[1];
    unsigned short* d = dst + fb * 32 + kk;
    d[0 * 32] = f2bf(a.x);
    d[1 * 32] = f2bf(a.y);
    d[2 * 32] = f2bf(a.z);
    d[3 * 32] = f2bf(a.w);
    d[4 * 32] = f2bf(b.x);
    d[5 * 32] = f2bf(b.y);
    d[6 * 32] = f2bf(b.z);
    d[7 * 32] = f2bf(b.w);
}

// --------------------------- h_acc[b,f] += sum_i v[b,i]*sum_j v[b,j]*W1
// grid.x : N tile of 64 columns (FF/64 = 16)
// grid.y : i-segment of ICHUNK i-values (512/8 = 64)
__global__ void __launch_bounds__(256)
bilinear_kernel(const float* __restrict__ W1,
                const unsigned short* __restrict__ vH,
                const float* __restrict__ vF,
                float* __restrict__ hacc) {
    extern __shared__ unsigned char smem[];
    unsigned short* sV = (unsigned short*)smem;                        // 128*512 bf16
    unsigned short* sB = (unsigned short*)(smem + BATCH * HIDDEN * 2); // 2 x 2048 bf16

    const int tid      = threadIdx.x;
    const int wave     = tid >> 5;
    const int lane     = tid & 31;
    const int laneRow  = lane & 15;
    const int laneHalf = lane >> 4;
    const int m0 = wave * 16;                 // this wave's 16 batch rows
    const int jb = laneHalf ? 16 : 0;         // A-fragment K base (contiguous k = 16h + s, matches B)
    const int kb = laneHalf ? 16 : 0;         // B-fragment K base (contiguous k = 16h + s, matches A)
    const int f0 = blockIdx.x * 64;
    const int i0 = blockIdx.y * ICHUNK;
    const size_t segBase = ((size_t)i0 * HIDDEN) * FF + f0;   // W1 offset of this segment

    // stage all of v (bf16, 128 KB) into LDS
    {
        const v4u* src = (const v4u*)vH;
        v4u*       dst = (v4u*)sV;
#pragma unroll
        for (int r = 0; r < 32; ++r)
            dst[tid + 256 * r] = src[tid + 256 * r];
    }

    // prime the tile pipeline: stage step 0 into buffer 0
    stage_tile(sB, W1, segBase, tid);

    v8f acc0 = {}, acc1 = {}, acc2 = {}, acc3 = {};
    v8f t0 = {}, t1 = {}, t2 = {}, t3 = {};
    int p = 0;

    const int NSTEP = ICHUNK * (HIDDEN / 32);   // 128 k-steps of 32
    for (int s = 0; s < NSTEP; ++s) {
        const int js = (s & 15) * 32;
        __syncthreads();   // buffer p now visible; buffer p^1 free for reuse

        if (s + 1 < NSTEP)
            stage_tile(sB + (p ^ 1) * BTILE, W1,
                       segBase + (size_t)(s + 1) * 32 * FF, tid);
        if (s + 2 < NSTEP)   // pull tile s+2 toward the caches (global_prefetch_b8)
            __builtin_prefetch(W1 + segBase + (size_t)(s + 2) * 32 * FF + (tid & 63), 0, 0);

        const unsigned short* bufp = sB + p * BTILE;

        // A fragment: raw v[b, js..js+31] in contiguous k = 16h + s order, same as B
        BfFrag A;
        const unsigned short* arow = sV + (m0 + laneRow) * HIDDEN + js + jb;
        A.q[0] = *(const v4u*)(arow);        // K = jb+0..jb+7
        A.q[1] = *(const v4u*)(arow + 8);    // K = jb+8..jb+15

        // B fragments: 4 column tiles of 16, K = kb..kb+15 contiguous in LDS
        BfFrag B0, B1, B2, B3;
        {
            const unsigned short* b0 = bufp + (0 * 16 + laneRow) * 32 + kb;
            const unsigned short* b1 = bufp + (1 * 16 + laneRow) * 32 + kb;
            const unsigned short* b2 = bufp + (2 * 16 + laneRow) * 32 + kb;
            const unsigned short* b3 = bufp + (3 * 16 + laneRow) * 32 + kb;
            B0.q[0] = *(const v4u*)(b0); B0.q[1] = *(const v4u*)(b0 + 8);
            B1.q[0] = *(const v4u*)(b1); B1.q[1] = *(const v4u*)(b1 + 8);
            B2.q[0] = *(const v4u*)(b2); B2.q[1] = *(const v4u*)(b2 + 8);
            B3.q[0] = *(const v4u*)(b3); B3.q[1] = *(const v4u*)(b3 + 8);
        }

        t0 = __builtin_amdgcn_wmma_f32_16x16x32_bf16(false, A.v, false, B0.v,
                                                     (short)0, t0, false, false);
        t1 = __builtin_amdgcn_wmma_f32_16x16x32_bf16(false, A.v, false, B1.v,
                                                     (short)0, t1, false, false);
        t2 = __builtin_amdgcn_wmma_f32_16x16x32_bf16(false, A.v, false, B2.v,
                                                     (short)0, t2, false, false);
        t3 = __builtin_amdgcn_wmma_f32_16x16x32_bf16(false, A.v, false, B3.v,
                                                     (short)0, t3, false, false);

        if ((s & 15) == 15) {
            // fold the outer-product factor v[b,i] in fp32
            const int i = i0 + (s >> 4);
#pragma unroll
            for (int r = 0; r < 8; ++r) {
                int   mm = m0 + (laneHalf ? 8 + r : r);
                float sc = vF[mm * HIDDEN + i];
                acc0[r] += sc * t0[r];
                acc1[r] += sc * t1[r];
                acc2[r] += sc * t2[r];
                acc3[r] += sc * t3[r];
            }
            t0 = (v8f){}; t1 = (v8f){}; t2 = (v8f){}; t3 = (v8f){};
        }
        p ^= 1;
    }

    // accumulate partials across i-segments
#pragma unroll
    for (int r = 0; r < 8; ++r) {
        int    mm  = m0 + (laneHalf ? 8 + r : r);
        float* row = hacc + (size_t)mm * FF + f0 + laneRow;
        __hip_atomic_fetch_add(row + 0,  acc0[r], __ATOMIC_RELAXED, __HIP_MEMORY_SCOPE_AGENT);
        __hip_atomic_fetch_add(row + 16, acc1[r], __ATOMIC_RELAXED, __HIP_MEMORY_SCOPE_AGENT);
        __hip_atomic_fetch_add(row + 32, acc2[r], __ATOMIC_RELAXED, __HIP_MEMORY_SCOPE_AGENT);
        __hip_atomic_fetch_add(row + 48, acc3[r], __ATOMIC_RELAXED, __HIP_MEMORY_SCOPE_AGENT);
    }
}

// ---------------------------------------------- h = gelu(h_acc + b1), exact erf
__global__ void __launch_bounds__(256)
gelu_kernel(const float* __restrict__ hacc, const float* __restrict__ b1,
            float* __restrict__ hg) {
    int idx = blockIdx.x * 256 + threadIdx.x;    // 131072 = 128*1024
    float x = hacc[idx] + b1[idx & (FF - 1)];
    hg[idx] = 0.5f * x * (1.0f + erff(x * 0.70710678118654752440f));
}

// ---------------------------------------------- out = h @ W2 + b2
__global__ void __launch_bounds__(256)
out_kernel(const float* __restrict__ hg, const float* __restrict__ W2,
           const float* __restrict__ b2, float* __restrict__ out) {
    int idx = blockIdx.x * 256 + threadIdx.x;    // 65536 = 128*512
    int b = idx >> 9;
    int o = idx & 511;
    const float* hr = hg + b * FF;
    float acc = b2[o];
#pragma unroll 8
    for (int f = 0; f < FF; ++f)
        acc += hr[f] * W2[f * OUTN + o];
    out[idx] = acc;
}

extern "C" void kernel_launch(void* const* d_in, const int* in_sizes, int n_in,
                              void* d_out, int out_size, void* d_ws, size_t ws_size,
                              hipStream_t stream) {
    const float* z  = (const float*)d_in[0];
    const float* Wv = (const float*)d_in[1];
    const float* bv = (const float*)d_in[2];
    const float* W1 = (const float*)d_in[3];
    const float* b1 = (const float*)d_in[4];
    const float* W2 = (const float*)d_in[5];
    const float* b2 = (const float*)d_in[6];
    float* out = (float*)d_out;

    char* ws = (char*)d_ws;
    float*          vF   = (float*)ws;                                  // 65536 f32
    unsigned short* vH   = (unsigned short*)(ws + 65536 * 4);           // 65536 bf16
    float*          hacc = (float*)(ws + 65536 * 4 + 65536 * 2);        // 131072 f32
    float*          hg   = (float*)(ws + 65536 * 4 + 65536 * 2 + 131072 * 4);

    zero_kernel<<<(131072 + 255) / 256, 256, 0, stream>>>(hacc, 131072);
    vproj_kernel<<<65536 / 256, 256, 0, stream>>>(z, Wv, bv, vF, vH);

    dim3 gB(FF / 64, HIDDEN / ICHUNK);                                  // 16 x 64 WGs
    size_t shmem = (size_t)BATCH * HIDDEN * 2 + 2 * BTILE * 2;          // 136 KB
    bilinear_kernel<<<gB, 256, shmem, stream>>>(W1, vH, vF, hacc);

    gelu_kernel<<<131072 / 256, 256, 0, stream>>>(hacc, b1, hg);
    out_kernel<<<65536 / 256, 256, 0, stream>>>(hg, W2, b2, out);
}
